// QuadraticSplineLayer_28870770164371
// MI455X (gfx1250) — matrix operand order repaired and checked by reading, not compiled, not run on hardware
//
#include <hip/hip_runtime.h>
#include <hip/hip_bf16.h>

// ---------------------------------------------------------------------------
// QuadraticSplineLayer for MI455X (gfx1250, wave32, WMMA bf16 matrix cores,
// Tensor Data Mover for GEMM operand staging)
//   B=8192, N=512, K=8, H=1024, out_dim = N*(2K+1) = 8704
// ---------------------------------------------------------------------------

#define BB 8192
#define NN 512
#define KK 8
#define HH 1024
#define OD 8704   // 512*17

typedef __bf16 bf16;
typedef __attribute__((ext_vector_type(16))) __bf16 v16bf;
typedef __attribute__((ext_vector_type(8)))  __bf16 v8bf;
typedef __attribute__((ext_vector_type(8)))  float  v8f;

#if __has_builtin(__builtin_amdgcn_tensor_load_to_lds) && \
    __has_builtin(__builtin_amdgcn_s_wait_tensorcnt)
#define USE_TDM 1
typedef __attribute__((ext_vector_type(4))) unsigned int u32x4;
typedef __attribute__((ext_vector_type(8))) int          i32x8;
typedef __attribute__((ext_vector_type(4))) int          i32x4;

// Runtime LDS byte offset of a shared pointer. The empty asm keeps the
// addrspacecast out of any constant/static initializer (ld.lld can't
// relocate those); the flat address's low 32 bits are the LDS byte offset
// (ISA 10.2: LDS aperture addresses truncate to addr[31:0]).
__device__ __forceinline__ unsigned int lds_byte_addr(void* p) {
    asm volatile("" : "+v"(p));
    return (unsigned int)(uintptr_t)p;
}

// Issue one TDM DMA: copy a (128 rows x 64 bytes) tile of a row-major 2-D
// tensor (row pitch = rowBytes) from global memory into LDS at lds_addr.
// D# per CDNA5 ISA ch.8: group0 = {flags, lds_addr, global_addr, type=2},
// group1 = dims/strides in 1-byte units, remaining groups zero (<=2-D tensor).
// This toolchain's builtin takes 6 args: (g0, g1, g2, g3, g4, cpol).
__device__ __forceinline__ void tdm_load_tile(unsigned int lds_addr,
                                              const bf16* gptr,
                                              unsigned int rowBytes)
{
    unsigned long long ga = (unsigned long long)(uintptr_t)gptr;
    u32x4 g0;
    g0[0] = 1u;                                           // count=1, user mode
    g0[1] = lds_addr;                                     // LDS byte address
    g0[2] = (unsigned int)ga;                             // global_addr[31:0]
    g0[3] = (unsigned int)((ga >> 32) & 0x1FFFFFFu)       // global_addr[56:32]
          | (2u << 30);                                   // type=2 ("image")
    unsigned int td0 = rowBytes;                          // tensor_dim0 (bytes)
    unsigned int td1 = 0xFFFFFFu;                         // tensor_dim1 (rows; large)
    i32x8 g1;
    g1[0] = 0;                                            // wg_mask=0, data_size=1B
    g1[1] = (int)((td0 & 0xFFFFu) << 16);                 // td0[15:0] @ bits 63:48
    g1[2] = (int)(((td0 >> 16) & 0xFFFFu)                 // td0[31:16]
          | ((td1 & 0xFFFFu) << 16));                     // td1[15:0]
    g1[3] = (int)(((td1 >> 16) & 0xFFFFu)                 // td1[31:16]
          | (64u << 16));                                 // tile_dim0 = 64 bytes
    g1[4] = 128;                                          // tile_dim1=128, tile_dim2=0
    g1[5] = (int)td0;                                     // dim0_stride[31:0] = rowBytes
    g1[6] = 0;                                            // dim0_stride[47:32]=0, dim1_stride lo
    g1[7] = 0;                                            // dim1_stride hi
    i32x4 z4 = {0, 0, 0, 0};
    i32x8 z8 = {0, 0, 0, 0, 0, 0, 0, 0};
    __builtin_amdgcn_tensor_load_to_lds(g0, g1, z4, z4, z8, 0);
}
#else
#define USE_TDM 0
#endif

// Branchless tanh on hardware transcendentals (v_exp_f32 + v_rcp_f32).
// Saturates correctly: x>>0 -> exp=inf -> 1; x<<0 -> exp=0 -> -1.
__device__ __forceinline__ float fast_tanh(float x) {
    float e = __expf(2.0f * x);
    return 1.0f - 2.0f * __builtin_amdgcn_rcpf(e + 1.0f);
}

// ---------------------------------------------------------------------------
// Weight transpose + f32 -> bf16 convert:  Wt[n][k] = bf16(W[k][n])
// ---------------------------------------------------------------------------
__global__ __launch_bounds__(256)
void convT_kernel(const float* __restrict__ W, bf16* __restrict__ Wt,
                  int Kd, int Nd)
{
    int i = blockIdx.x * 256 + threadIdx.x;
    if (i >= Kd * Nd) return;
    int k = i / Nd;
    int n = i - k * Nd;
    Wt[(size_t)n * Kd + k] = (bf16)W[i];
}

// A matrix for layer 1: bf16(v_passive - 0.5), row-major (B x N)
__global__ __launch_bounds__(256)
void convA_kernel(const float* __restrict__ v_in, bf16* __restrict__ Abf)
{
    int i = blockIdx.x * 256 + threadIdx.x;
    if (i >= BB * NN) return;
    int b = i >> 9;          // / 512
    int k = i & 511;
    Abf[i] = (bf16)(v_in[(size_t)b * (2 * NN) + k] - 0.5f);
}

// ---------------------------------------------------------------------------
// Fused GEMM + bias + tanh, bf16 in / bf16 out, f32 accumulate on WMMA.
//   C[M x Nd] = tanh(A[M x Kd] @ Bt^T + bias),  Bt is [Nd x Kd] (pre-transposed)
// Block tile 128x128, K-step 32, 8 waves (2x4), wave tile 64x32.
// Operand staging: TDM (tensor_load_to_lds) double-buffered; wave0 DMAs the
// A tile, wave1 the B tile, all waves sync on s_wait_tensorcnt + barrier.
// LDS map (bytes): A buf0 @0, A buf1 @8192, B buf0 @16384, B buf1 @24576.
// ---------------------------------------------------------------------------
__global__ __launch_bounds__(256)
void gemm_tanh_bf16_kernel(const bf16* __restrict__ A,
                           const bf16* __restrict__ Bt,
                           const float* __restrict__ bias,
                           bf16* __restrict__ C,
                           int M, int Kd, int Nd)
{
    // 32 KB: K-loop uses it as A[2][128*32] | B[2][128*32];
    // epilogue reuses the whole region as a 128x128 bf16 output tile.
    __shared__ bf16 smem[16384];

    const int tid  = threadIdx.x;
    const int lane = tid & 31;
    const int wave = tid >> 5;

    const int mBlk = blockIdx.y * 128;
    const int nBlk = blockIdx.x * 128;

    const int waveRow  = (wave >> 2) * 64;   // 0 or 64
    const int waveCol  = (wave & 3)  * 32;   // 0,32,64,96
    const int laneHalf = lane >> 4;          // 0 or 1
    const int lmod     = lane & 15;

    const bf16* aTile = A  + (size_t)mBlk * Kd;   // 128 rows, pitch Kd*2 B
    const bf16* bTile = Bt + (size_t)nBlk * Kd;

    v8f acc[4][2];
    #pragma unroll
    for (int i = 0; i < 4; ++i)
        #pragma unroll
        for (int j = 0; j < 2; ++j)
            acc[i][j] = v8f{};

    const unsigned int rowBytes = (unsigned int)(Kd * 2);
    const int nK = Kd >> 5;

#if USE_TDM
    const unsigned int ldsBase = lds_byte_addr(smem);
    // prologue: DMA K-slab 0 into buffer 0
    if (wave == 0)      tdm_load_tile(ldsBase, aTile, rowBytes);
    else if (wave == 1) tdm_load_tile(ldsBase + 16384u, bTile, rowBytes);
    if (wave < 2) __builtin_amdgcn_s_wait_tensorcnt(0);
    __syncthreads();
#else
    const int ldRow  = tid >> 1;          // 0..127
    const int ldHalf = (tid & 1) * 16;    // element offset 0 or 16
    const bf16* aSrc = aTile + (size_t)ldRow * Kd + ldHalf;
    const bf16* bSrc = bTile + (size_t)ldRow * Kd + ldHalf;
    uint4 aReg = *(const uint4*)aSrc;
    uint4 bReg = *(const uint4*)bSrc;
    *(uint4*)(smem + ldRow * 32 + ldHalf) = aReg;
    *(uint4*)(smem + 8192 + ldRow * 32 + ldHalf) = bReg;
    __syncthreads();
#endif

    for (int kt = 0; kt < nK; ++kt) {
        const int buf = kt & 1;
        const bf16* Ab = smem + (buf ? 4096 : 0);
        const bf16* Bb = smem + 8192 + (buf ? 4096 : 0);

#if USE_TDM
        // kick off the DMA of the next K-slab into the other buffer
        if (kt + 1 < nK) {
            unsigned int dstOff = (buf ^ 1) ? 8192u : 0u;
            if (wave == 0)
                tdm_load_tile(ldsBase + dstOff,
                              aTile + (size_t)(kt + 1) * 32, rowBytes);
            else if (wave == 1)
                tdm_load_tile(ldsBase + 16384u + dstOff,
                              bTile + (size_t)(kt + 1) * 32, rowBytes);
        }
#else
        if (kt + 1 < nK) {
            aReg = *(const uint4*)(aSrc + (size_t)(kt + 1) * 32);
            bReg = *(const uint4*)(bSrc + (size_t)(kt + 1) * 32);
        }
#endif

        // B fragments: lane<16 -> N=lmod, K=0..15 ; lane>=16 -> K=16..31
        v16bf bfrag[2];
        #pragma unroll
        for (int j = 0; j < 2; ++j) {
            int col = waveCol + j * 16 + lmod;
            bfrag[j] = *(const v16bf*)(Bb + col * 32 + laneHalf * 16);
        }

        #pragma unroll
        for (int i = 0; i < 4; ++i) {
            // A fragment: M=lmod row; K 0..15 (lane<16) or 16..31 (lane>=16)
            int m = waveRow + i * 16 + lmod;
            const bf16* ap = Ab + m * 32;
            v16bf afrag = *(const v16bf*)(ap + laneHalf * 16);
            #pragma unroll
            for (int j = 0; j < 2; ++j) {
                acc[i][j] = __builtin_amdgcn_wmma_f32_16x16x32_bf16(
                    false, afrag, false, bfrag[j],
                    (short)0, acc[i][j], false, false);
            }
        }

#if USE_TDM
        if (wave < 2) __builtin_amdgcn_s_wait_tensorcnt(0);
        __syncthreads();
#else
        __syncthreads();
        if (kt + 1 < nK) {
            int dstOff = (buf ^ 1) ? 4096 : 0;
            *(uint4*)(smem + dstOff + ldRow * 32 + ldHalf) = aReg;
            *(uint4*)(smem + 8192 + dstOff + ldRow * 32 + ldHalf) = bReg;
            __syncthreads();
        }
#endif
    }

    // ---- epilogue: bias + fast tanh -> bf16 tile in LDS ----
    // C tile layout: VGPR r -> lanes 0-15: M=r, N=lane ; lanes 16-31: M=8+r
    #pragma unroll
    for (int j = 0; j < 2; ++j) {
        int col = waveCol + j * 16 + lmod;
        float bv = bias[nBlk + col];
        #pragma unroll
        for (int i = 0; i < 4; ++i) {
            int mBase = waveRow + i * 16 + laneHalf * 8;
            #pragma unroll
            for (int r = 0; r < 8; ++r) {
                float v = fast_tanh(acc[i][j][r] + bv);
                smem[(mBase + r) * 128 + col] = (bf16)v;
            }
        }
    }
    __syncthreads();

    // ---- coalesced write-out: 128 rows x 256B via b128 stores ----
    {
        int row  = tid >> 1;             // 0..127
        int half = (tid & 1) * 8;        // uint4 chunk 0..7 or 8..15
        const uint4* src = (const uint4*)(smem + row * 128);
        uint4* dst = (uint4*)(C + (size_t)(mBlk + row) * Nd + nBlk);
        #pragma unroll
        for (int q = 0; q < 8; ++q)
            dst[half + q] = src[half + q];
    }
}

// ---------------------------------------------------------------------------
// Spline: one thread per (b, n). Reads 17 net values (bf16), produces
// v_out[b][n] (passive copy), v_out[b][N+n] (active), logterm[b*N+n].
// ---------------------------------------------------------------------------
__global__ __launch_bounds__(256)
void spline_kernel(const bf16* __restrict__ x3,
                   const float* __restrict__ v_in,
                   float* __restrict__ v_out,
                   float* __restrict__ logterm)
{
    int i = blockIdx.x * 256 + threadIdx.x;
    if (i >= BB * NN) return;
    int b = i >> 9;
    int n = i & 511;

    const bf16* p = x3 + (size_t)b * OD + n * 17;
    float net[17];
    #pragma unroll
    for (int j = 0; j < 17; ++j) net[j] = (float)p[j];
    const float* h = net;       // 9 heights
    const float* w = net + 9;   // 8 widths

    // softmax over widths
    float wmax = w[0];
    #pragma unroll
    for (int j = 1; j < 8; ++j) wmax = fmaxf(wmax, w[j]);
    float wn[8];
    float wsum = 0.f;
    #pragma unroll
    for (int j = 0; j < 8; ++j) { wn[j] = __expf(w[j] - wmax); wsum += wn[j]; }
    float winv = 1.f / wsum;
    #pragma unroll
    for (int j = 0; j < 8; ++j) wn[j] *= winv;

    float eh[9];
    #pragma unroll
    for (int j = 0; j < 9; ++j) eh[j] = __expf(h[j]);
    float denom = 0.f;
    #pragma unroll
    for (int j = 0; j < 8; ++j) denom += 0.5f * wn[j] * (eh[j] + eh[j + 1]);
    float dinv = 1.f / denom;
    float hn[9];
    #pragma unroll
    for (int j = 0; j < 9; ++j) hn[j] = eh[j] * dinv;

    float kx[9], ky[9];
    kx[0] = 0.f; ky[0] = 0.f;
    #pragma unroll
    for (int j = 0; j < 8; ++j) {
        kx[j + 1] = kx[j] + wn[j];
        ky[j + 1] = ky[j] + 0.5f * wn[j] * (hn[j] + hn[j + 1]);
    }

    float va = v_in[(size_t)b * (2 * NN) + NN + n];
    int cnt = 0;
    #pragma unroll
    for (int j = 0; j < 9; ++j) cnt += (kx[j] < va) ? 1 : 0;
    int k = min(max(cnt - 1, 0), KK - 1);

    float wseg = wn[k], hlo = hn[k], hhi = hn[k + 1];
    float xlo = kx[k], ylo = ky[k];
    float alpha = (va - xlo) / wseg;
    float out = ylo + alpha * hlo * wseg
              + 0.5f * alpha * alpha * (hhi - hlo) * wseg;

    v_out[(size_t)b * (2 * NN) + n]      = v_in[(size_t)b * (2 * NN) + n];
    v_out[(size_t)b * (2 * NN) + NN + n] = out;
    logterm[i] = __logf(hlo + alpha * (hhi - hlo));
}

// ---------------------------------------------------------------------------
// Per-batch-row reduction: log_density_out[b] = log_density[b] - sum_n logterm
// ---------------------------------------------------------------------------
__global__ __launch_bounds__(256)
void reduce_kernel(const float* __restrict__ logterm,
                   const float* __restrict__ log_density,
                   float* __restrict__ ld_out)
{
    __shared__ float s[256];
    int b = blockIdx.x;
    float sum = 0.f;
    for (int n = threadIdx.x; n < NN; n += 256)
        sum += logterm[(size_t)b * NN + n];
    s[threadIdx.x] = sum;
    __syncthreads();
    for (int off = 128; off > 0; off >>= 1) {
        if (threadIdx.x < off) s[threadIdx.x] += s[threadIdx.x + off];
        __syncthreads();
    }
    if (threadIdx.x == 0) ld_out[b] = log_density[b] - s[0];
}

// ---------------------------------------------------------------------------
extern "C" void kernel_launch(void* const* d_in, const int* in_sizes, int n_in,
                              void* d_out, int out_size, void* d_ws, size_t ws_size,
                              hipStream_t stream)
{
    const float* v_in  = (const float*)d_in[0];
    const float* logd  = (const float*)d_in[1];
    const float* W1    = (const float*)d_in[2];
    const float* b1    = (const float*)d_in[3];
    const float* W2    = (const float*)d_in[4];
    const float* b2    = (const float*)d_in[5];
    const float* W3    = (const float*)d_in[6];
    const float* b3    = (const float*)d_in[7];

    float* v_out  = (float*)d_out;                       // B x 2N
    float* ld_out = (float*)d_out + (size_t)BB * 2 * NN; // B x 1

    char*  base = (char*)d_ws;
    size_t off  = 0;
    auto alloc = [&](size_t bytes) -> char* {
        char* p = base + off;
        off += (bytes + 255) & ~(size_t)255;
        return p;
    };
    bf16*  W1t     = (bf16*)alloc((size_t)HH * NN * 2);
    bf16*  W2t     = (bf16*)alloc((size_t)HH * HH * 2);
    bf16*  W3t     = (bf16*)alloc((size_t)OD * HH * 2);
    bf16*  Abf     = (bf16*)alloc((size_t)BB * NN * 2);
    bf16*  x1bf    = (bf16*)alloc((size_t)BB * HH * 2);
    bf16*  x2bf    = (bf16*)alloc((size_t)BB * HH * 2);
    bf16*  x3bf    = (bf16*)alloc((size_t)BB * OD * 2);
    float* logterm = (float*)alloc((size_t)BB * NN * 4);
    (void)ws_size; (void)n_in; (void)in_sizes; (void)out_size;

    // 1) convert weights (transposed) + input activations to bf16
    convT_kernel<<<(NN * HH + 255) / 256, 256, 0, stream>>>(W1, W1t, NN, HH);
    convT_kernel<<<(HH * HH + 255) / 256, 256, 0, stream>>>(W2, W2t, HH, HH);
    convT_kernel<<<(HH * OD + 255) / 256, 256, 0, stream>>>(W3, W3t, HH, OD);
    convA_kernel<<<(BB * NN + 255) / 256, 256, 0, stream>>>(v_in, Abf);

    // 2) three WMMA GEMMs with fused bias+tanh (TDM-staged operands)
    dim3 blk(256);
    gemm_tanh_bf16_kernel<<<dim3(HH / 128, BB / 128), blk, 0, stream>>>(
        Abf, W1t, b1, x1bf, BB, NN, HH);
    gemm_tanh_bf16_kernel<<<dim3(HH / 128, BB / 128), blk, 0, stream>>>(
        x1bf, W2t, b2, x2bf, BB, HH, HH);
    gemm_tanh_bf16_kernel<<<dim3(OD / 128, BB / 128), blk, 0, stream>>>(
        x2bf, W3t, b3, x3bf, BB, HH, OD);

    // 3) spline transform + log-density reduction
    spline_kernel<<<(BB * NN + 255) / 256, 256, 0, stream>>>(
        x3bf, v_in, v_out, logterm);
    reduce_kernel<<<BB, 256, 0, stream>>>(logterm, logd, ld_out);
}
